// MixtralDecoderLayer_30262339567731
// MI455X (gfx1250) — matrix operand order repaired and checked by reading, not compiled, not run on hardware
//
#include <hip/hip_runtime.h>

// ---------- types ----------
typedef __bf16 bf16_t;
typedef bf16_t v16bf __attribute__((ext_vector_type(16)));
typedef float  v8f   __attribute__((ext_vector_type(8)));
typedef float  f4v   __attribute__((ext_vector_type(4)));

union AB16 { v16bf bf; f4v f4[2]; unsigned short us[16]; };

__device__ inline unsigned short f2bf(float f) {
  unsigned int u = __float_as_uint(f);
  u += 0x7FFFu + ((u >> 16) & 1u);   // round to nearest even
  return (unsigned short)(u >> 16);
}
__device__ inline float bf2f(unsigned short s) {
  return __uint_as_float(((unsigned int)s) << 16);
}

__device__ inline v8f wmma_bf16(const AB16& a, const AB16& b, v8f c) {
  // (neg_a, A, neg_b, B, c_mod, C, reuse_a, reuse_b)
  return __builtin_amdgcn_wmma_f32_16x16x32_bf16(false, a.bf, false, b.bf,
                                                 (short)0, c, false, false);
}

// scheduling-barrier waits: force load-group | wait | wmma-group ordering
#define WAIT_DS0()   asm volatile("s_wait_dscnt 0x0" ::: "memory")
#define WAIT_LD0()   asm volatile("s_wait_loadcnt 0x0" ::: "memory")
#define WAIT_LDDS0() asm volatile("s_wait_loadcnt_dscnt 0x0" ::: "memory")

// ---------- CDNA5 async global->LDS copy (ASYNCcnt-tracked) ----------
__device__ inline void async_load_b128(const unsigned short* lds_dst,
                                       const unsigned short* gsrc) {
  unsigned l = (unsigned)(size_t)lds_dst;  // low 32 bits of generic ptr = wave-relative LDS addr
  asm volatile("global_load_async_to_lds_b128 %0, %1, off"
               :: "v"(l), "v"(gsrc) : "memory");
}

// ---------- weight transpose + fp32 -> bf16 (src: K x N row-major f32; dst: N x K row-major bf16) ----------
__global__ void __launch_bounds__(256) transpose_cvt(const float* __restrict__ src,
                                                     unsigned short* __restrict__ dst,
                                                     int K, int N) {
  __shared__ unsigned short tile[32][33];
  size_t off = (size_t)blockIdx.z * (size_t)K * (size_t)N;
  src += off; dst += off;
  int n0 = blockIdx.x * 32, k0 = blockIdx.y * 32;
  int tx = threadIdx.x & 31;
  int ty = threadIdx.x >> 5;
#pragma unroll
  for (int i = 0; i < 4; i++) {
    int k = k0 + ty + i * 8;
    tile[ty + i * 8][tx] = f2bf(src[(size_t)k * N + n0 + tx]);
  }
  __syncthreads();
  int kk = threadIdx.x & 15;   // k-pair index
  int nn = threadIdx.x >> 4;   // 16 of 32 n values
#pragma unroll
  for (int i = 0; i < 2; i++) {
    int n = nn + i * 16;
    unsigned int val = (unsigned int)tile[2 * kk][n] |
                       ((unsigned int)tile[2 * kk + 1][n] << 16);
    *(unsigned int*)(dst + (size_t)(n0 + n) * K + k0 + 2 * kk) = val;
  }
}

// ---------- rmsnorm: f32 in -> bf16 out (+ optional raw f32 passthrough copy) ----------
__global__ void __launch_bounds__(256) rmsnorm_kernel(const float* __restrict__ x,
                                                      const float* __restrict__ w,
                                                      unsigned short* __restrict__ o,
                                                      float* __restrict__ passthru) {
  int row = blockIdx.x, tid = threadIdx.x;
  int lane = tid & 31, wid = tid >> 5;
  const float* xr = x + (size_t)row * 1024;
  f4v v = *(const f4v*)(xr + tid * 4);
  float ss = v[0] * v[0] + v[1] * v[1] + v[2] * v[2] + v[3] * v[3];
#pragma unroll
  for (int m = 16; m >= 1; m >>= 1) ss += __shfl_xor(ss, m);
  __shared__ float red[8];
  if (lane == 0) red[wid] = ss;
  __syncthreads();
  float tot = 0.f;
#pragma unroll
  for (int i = 0; i < 8; i++) tot += red[i];
  float sc = rsqrtf(tot * (1.0f / 1024.0f) + 1e-5f);
  f4v wv4 = *(const f4v*)(w + tid * 4);
#pragma unroll
  for (int i = 0; i < 4; i++) {
    size_t idx = (size_t)row * 1024 + tid * 4 + i;
    o[idx] = f2bf(v[i] * sc * wv4[i]);
    if (passthru) passthru[idx] = v[i];
  }
}

// ---------- generic WMMA GEMM with async-LDS B staging ----------
// A[MxK] bf16 row-major (direct global), Bt[NxK] bf16 row-major (staged in LDS, 8x reuse)
// EPI: 0 = C f32, 1 = C = acc + R (residual), 2 = C += scale[m*8+expert]*acc
template <int EPI>
__global__ void __launch_bounds__(256) gemm_tn(const unsigned short* __restrict__ A,
                                               const unsigned short* __restrict__ Bt,
                                               float* __restrict__ C,
                                               const float* __restrict__ R,
                                               const float* __restrict__ scale, int expert,
                                               int M, int N, int K) {
  __shared__ __align__(16) unsigned short btile[2][64 * 32];  // [buf][n(64) x k(32)] bf16
  int tid = threadIdx.x;
  int lane = tid & 31, wid = tid >> 5;
  int half = lane >> 4, rsel = lane & 15;
  int m0 = blockIdx.y * 128 + wid * 16;
  int n0 = blockIdx.x * 64;

  int brow = tid >> 2, bch = tid & 3;
  const unsigned short* bsrc = Bt + (size_t)(n0 + brow) * K + bch * 8;
  unsigned short* bdst0 = &btile[0][brow * 32 + bch * 8];
  unsigned short* bdst1 = &btile[1][brow * 32 + bch * 8];

  async_load_b128(bdst0, bsrc);  // prefetch k=0

  v8f acc[4] = {};
  const unsigned short* ap = A + (size_t)(m0 + rsel) * K + half * 8;

  auto mm = [&](const unsigned short* bb, const AB16& a) {
    AB16 bfr[4];
#pragma unroll
    for (int g = 0; g < 4; g++) {
      const unsigned short* bp = bb + (g * 16 + rsel) * 32 + half * 8;
      bfr[g].f4[0] = *(const f4v*)bp;
      bfr[g].f4[1] = *(const f4v*)(bp + 16);
    }
    WAIT_DS0();  // sched barrier: cluster the 8 ds_loads, then back-to-back WMMAs
#pragma unroll
    for (int g = 0; g < 4; g++) acc[g] = wmma_bf16(a, bfr[g], acc[g]);
  };

  int KT = K >> 5;
  for (int kt = 0; kt < KT - 1; ++kt) {  // steady state: always prefetch, wait<=1
    int k = kt << 5;
    int cur = kt & 1;
    AB16 a;  // issue A loads first: overlap the async-wait + barrier
    a.f4[0] = *(const f4v*)(ap + k);
    a.f4[1] = *(const f4v*)(ap + k + 16);
    async_load_b128(cur ? bdst0 : bdst1, bsrc + k + 32);
    asm volatile("s_wait_asynccnt 0x1" ::: "memory");
    __syncthreads();
    mm(&btile[cur][0], a);
    __syncthreads();
  }
  {  // peeled last tile: no prefetch, drain ASYNCcnt
    int k = (KT - 1) << 5;
    int cur = (KT - 1) & 1;
    AB16 a;
    a.f4[0] = *(const f4v*)(ap + k);
    a.f4[1] = *(const f4v*)(ap + k + 16);
    asm volatile("s_wait_asynccnt 0x0" ::: "memory");
    __syncthreads();
    mm(&btile[cur][0], a);
  }

  if (EPI == 2) {
    float sc[8];
#pragma unroll
    for (int r = 0; r < 8; r++) sc[r] = scale[(size_t)(m0 + r + 8 * half) * 8 + expert];
#pragma unroll
    for (int g = 0; g < 4; g++)
#pragma unroll
      for (int r = 0; r < 8; r++) {
        size_t idx = (size_t)(m0 + r + 8 * half) * N + n0 + g * 16 + rsel;
        C[idx] += sc[r] * acc[g][r];
      }
  } else {
#pragma unroll
    for (int g = 0; g < 4; g++)
#pragma unroll
      for (int r = 0; r < 8; r++) {
        size_t idx = (size_t)(m0 + r + 8 * half) * N + n0 + g * 16 + rsel;
        float v = acc[g][r];
        if (EPI == 1) v += R[idx];
        C[idx] = v;
      }
  }
}

// ---------- fused MoE stage 1: Out = bf16( silu(A@B1) * (A@B2) ), dual async-LDS staging ----------
__global__ void __launch_bounds__(256) gemm_dual_silu(const unsigned short* __restrict__ A,
                                                      const unsigned short* __restrict__ B1t,
                                                      const unsigned short* __restrict__ B2t,
                                                      unsigned short* __restrict__ Out,
                                                      int M, int N, int K) {
  __shared__ __align__(16) unsigned short b1s[2][64 * 32];
  __shared__ __align__(16) unsigned short b2s[2][64 * 32];
  int tid = threadIdx.x;
  int lane = tid & 31, wid = tid >> 5;
  int half = lane >> 4, rsel = lane & 15;
  int m0 = blockIdx.y * 128 + wid * 16;
  int n0 = blockIdx.x * 64;

  int brow = tid >> 2, bch = tid & 3;
  size_t soff = (size_t)(n0 + brow) * K + bch * 8;
  const unsigned short* s1 = B1t + soff;
  const unsigned short* s2 = B2t + soff;
  unsigned short* d10 = &b1s[0][brow * 32 + bch * 8];
  unsigned short* d11 = &b1s[1][brow * 32 + bch * 8];
  unsigned short* d20 = &b2s[0][brow * 32 + bch * 8];
  unsigned short* d21 = &b2s[1][brow * 32 + bch * 8];

  async_load_b128(d10, s1);
  async_load_b128(d20, s2);

  v8f acg[4] = {};
  v8f acv[4] = {};
  const unsigned short* ap = A + (size_t)(m0 + rsel) * K + half * 8;

  auto mm2 = [&](const unsigned short* bb1, const unsigned short* bb2, const AB16& a) {
    AB16 f1[4], f2[4];
#pragma unroll
    for (int g = 0; g < 4; g++) {
      int fo = (g * 16 + rsel) * 32 + half * 8;
      f1[g].f4[0] = *(const f4v*)(bb1 + fo);
      f1[g].f4[1] = *(const f4v*)(bb1 + fo + 16);
      f2[g].f4[0] = *(const f4v*)(bb2 + fo);
      f2[g].f4[1] = *(const f4v*)(bb2 + fo + 16);
    }
    WAIT_DS0();  // cluster all 16 ds_loads, then 8 back-to-back WMMAs
#pragma unroll
    for (int g = 0; g < 4; g++) acg[g] = wmma_bf16(a, f1[g], acg[g]);
#pragma unroll
    for (int g = 0; g < 4; g++) acv[g] = wmma_bf16(a, f2[g], acv[g]);
  };

  int KT = K >> 5;
  for (int kt = 0; kt < KT - 1; ++kt) {
    int k = kt << 5;
    int cur = kt & 1;
    AB16 a;
    a.f4[0] = *(const f4v*)(ap + k);
    a.f4[1] = *(const f4v*)(ap + k + 16);
    async_load_b128(cur ? d10 : d11, s1 + k + 32);
    async_load_b128(cur ? d20 : d21, s2 + k + 32);
    asm volatile("s_wait_asynccnt 0x2" ::: "memory");
    __syncthreads();
    mm2(&b1s[cur][0], &b2s[cur][0], a);
    __syncthreads();
  }
  {
    int k = (KT - 1) << 5;
    int cur = (KT - 1) & 1;
    AB16 a;
    a.f4[0] = *(const f4v*)(ap + k);
    a.f4[1] = *(const f4v*)(ap + k + 16);
    asm volatile("s_wait_asynccnt 0x0" ::: "memory");
    __syncthreads();
    mm2(&b1s[cur][0], &b2s[cur][0], a);
  }
#pragma unroll
  for (int g = 0; g < 4; g++)
#pragma unroll
    for (int r = 0; r < 8; r++) {
      float gv = acg[g][r];
      float sv = gv / (1.0f + __expf(-gv)) * acv[g][r];
      Out[(size_t)(m0 + r + 8 * half) * N + n0 + g * 16 + rsel] = f2bf(sv);
    }
}

// ---------- RoPE + repack to attention-friendly layouts ----------
__global__ void __launch_bounds__(256) rope_pack(const float* __restrict__ qf,
                                                 const float* __restrict__ kf,
                                                 const float* __restrict__ vf,
                                                 const float* __restrict__ cosb,
                                                 const float* __restrict__ sinb,
                                                 unsigned short* __restrict__ qbf,
                                                 unsigned short* __restrict__ kbf,
                                                 unsigned short* __restrict__ vtbf) {
  int idx = blockIdx.x * blockDim.x + threadIdx.x;  // 0 .. 2*1024*16*64-1
  {
    int d = idx & 63, h = (idx >> 6) & 15, s = (idx >> 10) & 1023, b = idx >> 20;
    const float* base = qf + ((size_t)(b * 1024 + s)) * 1024 + h * 64;
    int j = d & 31;
    float x1 = base[2 * j], x2 = base[2 * j + 1];
    float c = cosb[s * 32 + j], sn = sinb[s * 32 + j];
    float o = (d < 32) ? (x1 * c - x2 * sn) : (x1 * sn + x2 * c);
    qbf[((size_t)(b * 16 + h) * 1024 + s) * 64 + d] = f2bf(o);
  }
  if (idx < 2 * 4 * 1024 * 64) {
    int d = idx & 63, h = (idx >> 6) & 3, s = (idx >> 8) & 1023, b = idx >> 18;
    const float* base = kf + ((size_t)(b * 1024 + s)) * 256 + h * 64;
    int j = d & 31;
    float x1 = base[2 * j], x2 = base[2 * j + 1];
    float c = cosb[s * 32 + j], sn = sinb[s * 32 + j];
    float o = (d < 32) ? (x1 * c - x2 * sn) : (x1 * sn + x2 * c);
    kbf[((size_t)(b * 4 + h) * 1024 + s) * 64 + d] = f2bf(o);
    float vv = vf[((size_t)(b * 1024 + s)) * 256 + h * 64 + d];
    vtbf[((size_t)(b * 4 + h) * 64 + d) * 1024 + s] = f2bf(vv);
  }
}

// ---------- flash attention, one wave = 16 query rows of one head ----------
__global__ void __launch_bounds__(256) attn_kernel(const unsigned short* __restrict__ qbf,
                                                   const unsigned short* __restrict__ kbf,
                                                   const unsigned short* __restrict__ vtbf,
                                                   unsigned short* __restrict__ obf) {
  __shared__ __align__(16) unsigned short pshm[8][512];  // per-wave 16x32 bf16 P tile
  int lane = threadIdx.x & 31, wid = threadIdx.x >> 5;
  int w = blockIdx.x * 8 + wid;
  int qt = w & 63, h = (w >> 6) & 15, b = w >> 10;
  int kvh = h >> 2, half = lane >> 4, n = lane & 15;

  AB16 qa0, qa1;  // Q A-fragments, d=0..31 and d=32..63
  const unsigned short* qp = qbf + ((size_t)(b * 16 + h) * 1024 + qt * 16 + n) * 64;
  qa0.f4[0] = *(const f4v*)(qp + half * 8);
  qa0.f4[1] = *(const f4v*)(qp + 16 + half * 8);
  qa1.f4[0] = *(const f4v*)(qp + 32 + half * 8);
  qa1.f4[1] = *(const f4v*)(qp + 48 + half * 8);

  v8f oacc[4] = {};
  float mrun[8], lrun[8], cs[8];
#pragma unroll
  for (int r = 0; r < 8; r++) { mrun[r] = -3e38f; lrun[r] = 0.f; }

  const unsigned short* kb = kbf + (size_t)(b * 4 + kvh) * 65536;
  const unsigned short* vb = vtbf + (size_t)(b * 4 + kvh) * 65536;
  int qmax = qt * 16 + 15;

  for (int kc = 0; kc <= qmax; kc += 32) {
    v8f s0 = {}, s1 = {};
    {
      const unsigned short* kp = kb + (size_t)(kc + n) * 64 + half * 8;
      AB16 kf4[4];
      kf4[0].f4[0] = *(const f4v*)(kp);
      kf4[0].f4[1] = *(const f4v*)(kp + 16);
      kf4[1].f4[0] = *(const f4v*)(kp + 32);
      kf4[1].f4[1] = *(const f4v*)(kp + 48);
      kf4[2].f4[0] = *(const f4v*)(kp + 1024);
      kf4[2].f4[1] = *(const f4v*)(kp + 1040);
      kf4[3].f4[0] = *(const f4v*)(kp + 1056);
      kf4[3].f4[1] = *(const f4v*)(kp + 1072);
      WAIT_LD0();  // cluster the 8 global_load_b128, then 4 WMMAs
      s0 = wmma_bf16(qa0, kf4[0], s0);
      s0 = wmma_bf16(qa1, kf4[1], s0);
      s1 = wmma_bf16(qa0, kf4[2], s1);
      s1 = wmma_bf16(qa1, kf4[3], s1);
    }
#pragma unroll
    for (int r = 0; r < 8; r++) {
      int row = qt * 16 + r + 8 * half;
      float a0 = s0[r] * 0.125f + ((kc + n) <= row ? 0.f : -1e9f);
      float a1 = s1[r] * 0.125f + ((kc + 16 + n) <= row ? 0.f : -1e9f);
      float lm = fmaxf(a0, a1);
#pragma unroll
      for (int m = 1; m < 16; m <<= 1) lm = fmaxf(lm, __shfl_xor(lm, m));
      float nm = fmaxf(mrun[r], lm);
      float p0 = __expf(a0 - nm), p1 = __expf(a1 - nm);
      float rs = p0 + p1;
#pragma unroll
      for (int m = 1; m < 16; m <<= 1) rs += __shfl_xor(rs, m);
      cs[r] = __expf(mrun[r] - nm);
      lrun[r] = lrun[r] * cs[r] + rs;
      mrun[r] = nm;
      int rr = r + 8 * half;
      pshm[wid][rr * 32 + n] = f2bf(p0);
      pshm[wid][rr * 32 + 16 + n] = f2bf(p1);
    }
#pragma unroll
    for (int g = 0; g < 4; g++)
#pragma unroll
      for (int r = 0; r < 8; r++) oacc[g][r] *= cs[r];
    WAIT_DS0();  // wave-local LDS ordering (waves diverge; no block barrier)
    AB16 pa;
    const unsigned short* pp = &pshm[wid][n * 32 + half * 8];
    pa.f4[0] = *(const f4v*)pp;
    pa.f4[1] = *(const f4v*)(pp + 16);
    AB16 vf4[4];
#pragma unroll
    for (int g = 0; g < 4; g++) {
      const unsigned short* vp = vb + (size_t)(g * 16 + n) * 1024 + kc + half * 8;
      vf4[g].f4[0] = *(const f4v*)vp;
      vf4[g].f4[1] = *(const f4v*)(vp + 16);
    }
    WAIT_LDDS0();  // P(LDS) + V(global) clustered, then 4 WMMAs
#pragma unroll
    for (int g = 0; g < 4; g++) oacc[g] = wmma_bf16(pa, vf4[g], oacc[g]);
  }
#pragma unroll
  for (int g = 0; g < 4; g++)
#pragma unroll
    for (int r = 0; r < 8; r++) {
      int s = qt * 16 + r + 8 * half;
      obf[(size_t)(b * 1024 + s) * 1024 + h * 64 + g * 16 + n] = f2bf(oacc[g][r] / lrun[r]);
    }
}

// ---------- router: logits = x2 @ gate_w, top-2 softmax -> dense we[token][8] ----------
__global__ void __launch_bounds__(256) router_kernel(const unsigned short* __restrict__ x2,
                                                     const float* __restrict__ gate,
                                                     float* __restrict__ we) {
  int lane = threadIdx.x & 31;
  int tok = blockIdx.x * 8 + (threadIdx.x >> 5);
  float acc[8] = {0, 0, 0, 0, 0, 0, 0, 0};
  const unsigned short* xr = x2 + (size_t)tok * 1024;
  for (int hh = lane; hh < 1024; hh += 32) {
    float xv = bf2f(xr[hh]);
    const float* gr = gate + (size_t)hh * 8;
#pragma unroll
    for (int e = 0; e < 8; e++) acc[e] += xv * gr[e];
  }
#pragma unroll
  for (int e = 0; e < 8; e++)
#pragma unroll
    for (int m = 16; m >= 1; m >>= 1) acc[e] += __shfl_xor(acc[e], m);
  int i1 = 0;
  float b1 = acc[0];
  for (int e = 1; e < 8; e++)
    if (acc[e] > b1) { b1 = acc[e]; i1 = e; }
  int i2 = -1;
  float b2 = -3e38f;
  for (int e = 0; e < 8; e++)
    if (e != i1 && acc[e] > b2) { b2 = acc[e]; i2 = e; }
  float p1 = 1.0f / (1.0f + __expf(b2 - b1));
  float p2 = 1.0f / (1.0f + __expf(b1 - b2));
  if (lane < 8) we[(size_t)tok * 8 + lane] = (lane == i1) ? p1 : ((lane == i2) ? p2 : 0.0f);
}

// ---------- host ----------
extern "C" void kernel_launch(void* const* d_in, const int* in_sizes, int n_in,
                              void* d_out, int out_size, void* d_ws, size_t ws_size,
                              hipStream_t stream) {
  (void)in_sizes; (void)n_in; (void)out_size; (void)ws_size;
  const float* hidden = (const float*)d_in[0];
  // d_in[1] = attention_mask (causal, recomputed on device)
  const float* norm1 = (const float*)d_in[2];
  const float* norm2 = (const float*)d_in[3];
  const float* wq = (const float*)d_in[4];
  const float* wk = (const float*)d_in[5];
  const float* wv = (const float*)d_in[6];
  const float* wo = (const float*)d_in[7];
  const float* gate = (const float*)d_in[8];
  const float* w1 = (const float*)d_in[9];
  const float* vw = (const float*)d_in[10];
  const float* w2 = (const float*)d_in[11];
  const float* cosb = (const float*)d_in[12];
  const float* sinb = (const float*)d_in[13];
  float* out = (float*)d_out;

  char* p = (char*)d_ws;
  auto carve = [&](size_t bytes) -> char* {
    char* r = p;
    p += (bytes + 255) & ~(size_t)255;
    return r;
  };
  unsigned short* wqT = (unsigned short*)carve(1024ull * 1024 * 2);
  unsigned short* wkT = (unsigned short*)carve(256ull * 1024 * 2);
  unsigned short* wvT = (unsigned short*)carve(256ull * 1024 * 2);
  unsigned short* woT = (unsigned short*)carve(1024ull * 1024 * 2);
  unsigned short* w1T = (unsigned short*)carve(8ull * 3584 * 1024 * 2);
  unsigned short* vwT = (unsigned short*)carve(8ull * 3584 * 1024 * 2);
  unsigned short* w2T = (unsigned short*)carve(8ull * 1024 * 3584 * 2);
  unsigned short* x1bf = (unsigned short*)carve(2048ull * 1024 * 2);
  float* qf32 = (float*)carve(2048ull * 1024 * 4);
  float* kf32 = (float*)carve(2048ull * 256 * 4);
  float* vf32 = (float*)carve(2048ull * 256 * 4);
  unsigned short* qbf = (unsigned short*)carve(2048ull * 1024 * 2);
  unsigned short* kbfp = (unsigned short*)carve(524288ull * 2);
  unsigned short* vtbf = (unsigned short*)carve(524288ull * 2);
  unsigned short* obf = (unsigned short*)carve(2048ull * 1024 * 2);
  float* hbuf = (float*)carve(2048ull * 1024 * 4);
  unsigned short* x2bf = (unsigned short*)carve(2048ull * 1024 * 2);
  float* webuf = (float*)carve(2048ull * 8 * 4);
  unsigned short* ffmid = (unsigned short*)carve(2048ull * 3584 * 2);

  transpose_cvt<<<dim3(32, 32, 1), 256, 0, stream>>>(wq, wqT, 1024, 1024);
  transpose_cvt<<<dim3(8, 32, 1), 256, 0, stream>>>(wk, wkT, 1024, 256);
  transpose_cvt<<<dim3(8, 32, 1), 256, 0, stream>>>(wv, wvT, 1024, 256);
  transpose_cvt<<<dim3(32, 32, 1), 256, 0, stream>>>(wo, woT, 1024, 1024);
  transpose_cvt<<<dim3(112, 32, 8), 256, 0, stream>>>(w1, w1T, 1024, 3584);
  transpose_cvt<<<dim3(112, 32, 8), 256, 0, stream>>>(vw, vwT, 1024, 3584);
  transpose_cvt<<<dim3(32, 112, 8), 256, 0, stream>>>(w2, w2T, 3584, 1024);

  rmsnorm_kernel<<<2048, 256, 0, stream>>>(hidden, norm1, x1bf, nullptr);

  gemm_tn<0><<<dim3(16, 16), 256, 0, stream>>>(x1bf, wqT, qf32, nullptr, nullptr, 0, 2048, 1024, 1024);
  gemm_tn<0><<<dim3(4, 16), 256, 0, stream>>>(x1bf, wkT, kf32, nullptr, nullptr, 0, 2048, 256, 1024);
  gemm_tn<0><<<dim3(4, 16), 256, 0, stream>>>(x1bf, wvT, vf32, nullptr, nullptr, 0, 2048, 256, 1024);

  rope_pack<<<8192, 256, 0, stream>>>(qf32, kf32, vf32, cosb, sinb, qbf, kbfp, vtbf);

  attn_kernel<<<256, 256, 0, stream>>>(qbf, kbfp, vtbf, obf);

  gemm_tn<1><<<dim3(16, 16), 256, 0, stream>>>(obf, woT, hbuf, hidden, nullptr, 0, 2048, 1024, 1024);

  rmsnorm_kernel<<<2048, 256, 0, stream>>>(hbuf, norm2, x2bf, out);  // also copies h into d_out

  router_kernel<<<256, 256, 0, stream>>>(x2bf, gate, webuf);

  for (int e = 0; e < 8; e++) {
    gemm_dual_silu<<<dim3(56, 16), 256, 0, stream>>>(x2bf, w1T + (size_t)e * 3584 * 1024,
                                                     vwT + (size_t)e * 3584 * 1024, ffmid,
                                                     2048, 3584, 1024);
    gemm_tn<2><<<dim3(16, 16), 256, 0, stream>>>(ffmid, w2T + (size_t)e * 1024 * 3584, out,
                                                 nullptr, webuf, e, 2048, 1024, 3584);
  }
}
